// FermiNet_63196148793908
// MI455X (gfx1250) — matrix operand order repaired and checked by reading, not compiled, not run on hardware
//
#include <hip/hip_runtime.h>
#include <hip/hip_bf16.h>
#include <math.h>

// N=1024, DIM=3, DEPTH=4, H1=64, H2=32, NF=8, L=1
// Workspace layout (floats):
//   partial : 16*1024*96 = 1,572,864   (per-iblock g2 partial sums for h2_1..h2_3)
//   g2_0    : 1024*49    = 50,176      (mean_i of raw 49-dim pair features)
//   g2m     : 1024*96    = 98,304      (mean_i of h2_1, h2_2, h2_3; 32 each)
//   g1      : 64                        (mean over rows of current h1)
//   h1a/h1b : 1024*64 each              (ping-pong h1 buffers)
// Total ~7.1 MB.

typedef __attribute__((ext_vector_type(16))) _Float16 v16h;
typedef __attribute__((ext_vector_type(8)))  _Float16 v8h;
typedef __attribute__((ext_vector_type(8)))  float    v8f;

#define PI_F     3.14159265358979323846f
#define TWO_PI_F 6.28318530717958647692f

__device__ __forceinline__ float ftanh(float x) {
    float ax = fabsf(x);
    float e  = __expf(-2.0f * ax);          // v_exp_f32
    float r  = (1.0f - e) / (1.0f + e);
    return copysignf(r, x);
}

// 49 pair features: [r, {cos,sin blocks per freq k=1..8, dim-major inside}]
__device__ __forceinline__ void pair_features(const float rij0, const float rij1,
                                              const float rij2, float ft[49]) {
    float s0 = __sinf(PI_F * rij0);
    float s1 = __sinf(PI_F * rij1);
    float s2 = __sinf(PI_F * rij2);
    ft[0] = sqrtf(s0 * s0 + s1 * s1 + s2 * s2);
    float rij[3] = {rij0, rij1, rij2};
#pragma unroll
    for (int kf = 1; kf <= 8; ++kf) {
#pragma unroll
        for (int d = 0; d < 3; ++d) {
            float w = (TWO_PI_F * (float)kf) * rij[d];
            ft[1 + (kf - 1) * 6 + d]     = __cosf(w);
            ft[1 + (kf - 1) * 6 + 3 + d] = __sinf(w);
        }
    }
}

// Sum D over rows m (16 pairs): per-lane sum of 8 VGPRs covers m = 8*lhalf..+7,
// then xor-16 shuffle folds the other half. Lane `lane` owns feature n = lane.
__device__ __forceinline__ float reduce_mn(const v8f a, const v8f b, int lhalf) {
    float sA = 0.f, sB = 0.f;
#pragma unroll
    for (int r = 0; r < 8; ++r) { sA += a[r]; sB += b[r]; }
    sA += __shfl_xor(sA, 16);
    sB += __shfl_xor(sB, 16);
    return lhalf ? sB : sA;
}

__device__ __forceinline__ void lds_store_tile(_Float16* tile, const v8f a, const v8f b,
                                               int lhalf, int lm) {
#pragma unroll
    for (int r = 0; r < 8; ++r) {
        int m = lhalf * 8 + r;                  // D layout: VGPR r -> row m
        tile[m * 40 + lm]      = (_Float16)a[r];   // n = lm        (tile 0)
        tile[m * 40 + 16 + lm] = (_Float16)b[r];   // n = lm + 16   (tile 1)
    }
}

__device__ __forceinline__ v16h lds_load_A(const _Float16* tile, int lhalf, int lm) {
    // A layout f16 16x32: lane<16: K = {0..7,16..23}; lane>=16: +8
    const v8h* p0 = reinterpret_cast<const v8h*>(tile + lm * 40 + 8 * lhalf);
    const v8h* p1 = reinterpret_cast<const v8h*>(tile + lm * 40 + 16 + 8 * lhalf);
    v8h lo = *p0, hi = *p1;
    v16h a;
#pragma unroll
    for (int h = 0; h < 8; ++h) { a[h] = lo[h]; a[h + 8] = hi[h]; }
    return a;
}

// ---------------------------------------------------------------------------
// Kernel: mean over i of raw 49-dim features -> g2_0[j][49]   (one block per j)
// ---------------------------------------------------------------------------
__global__ __launch_bounds__(256) void g2_0_kernel(const float* __restrict__ x,
                                                   float* __restrict__ g2_0) {
    int j = blockIdx.x;
    int tid = threadIdx.x, wave = tid >> 5, lane = tid & 31;
    float xj0 = x[j * 3 + 0], xj1 = x[j * 3 + 1], xj2 = x[j * 3 + 2];
    float acc[49];
#pragma unroll
    for (int k = 0; k < 49; ++k) acc[k] = 0.f;
    for (int i = tid; i < 1024; i += 256) {
        float ft[49];
        pair_features(x[i * 3 + 0] - xj0, x[i * 3 + 1] - xj1, x[i * 3 + 2] - xj2, ft);
#pragma unroll
        for (int k = 0; k < 49; ++k) acc[k] += ft[k];
    }
#pragma unroll
    for (int off = 16; off; off >>= 1)
#pragma unroll
        for (int k = 0; k < 49; ++k) acc[k] += __shfl_xor(acc[k], off);
    __shared__ float ws[8][49];
    if (lane == 0) {
#pragma unroll
        for (int k = 0; k < 49; ++k) ws[wave][k] = acc[k];
    }
    __syncthreads();
    if (tid < 49) {
        float s = 0.f;
#pragma unroll
        for (int w = 0; w < 8; ++w) s += ws[w][tid];
        g2_0[j * 49 + tid] = s * (1.0f / 1024.0f);
    }
}

// ---------------------------------------------------------------------------
// Main fused h2 kernel: features -> 3 WMMA layers with tanh+residual, all in
// registers/LDS; emits per-iblock column-sum partials for g2_1..g2_3.
// Grid (16 jb, 16 ib), 256 threads (8 waves). Wave w owns j_local = 8w..8w+7.
// ---------------------------------------------------------------------------
__global__ __launch_bounds__(256) void h2_main(
    const float* __restrict__ x,
    const float* __restrict__ W2_0, const float* __restrict__ b2_0,
    const float* __restrict__ W2_1, const float* __restrict__ b2_1,
    const float* __restrict__ W2_2, const float* __restrict__ b2_2,
    float* __restrict__ partial /* [16][1024][96] */) {
    const int jb = blockIdx.x, ib = blockIdx.y;
    const int tid = threadIdx.x;
    const int wave = tid >> 5, lane = tid & 31;
    const int lhalf = lane >> 4, lm = lane & 15;

    __shared__ __align__(16) _Float16 xT[8][16 * 40];  // per-wave transpose tile

    // Preload B operands (f16). B layout: lane<16: K = {0..7,16..23}; lane>=16: +8.
    v16h B0[2][2], B1[2], B2[2];
#pragma unroll
    for (int t = 0; t < 2; ++t) {
        int n = lm + 16 * t;
#pragma unroll
        for (int c = 0; c < 2; ++c) {
#pragma unroll
            for (int h = 0; h < 16; ++h) {
                int K = c * 32 + lhalf * 8 + ((h < 8) ? h : (h + 8));
                B0[t][c][h] = (_Float16)((K < 49) ? W2_0[K * 32 + n] : 0.0f);
            }
        }
#pragma unroll
        for (int h = 0; h < 16; ++h) {
            int K = lhalf * 8 + ((h < 8) ? h : (h + 8));
            B1[t][h] = (_Float16)W2_1[K * 32 + n];
            B2[t][h] = (_Float16)W2_2[K * 32 + n];
        }
    }
    const float bia0[2] = {b2_0[lm], b2_0[lm + 16]};
    const float bia1[2] = {b2_1[lm], b2_1[lm + 16]};
    const float bia2[2] = {b2_2[lm], b2_2[lm + 16]};
    const v8f zero = {0.f, 0.f, 0.f, 0.f, 0.f, 0.f, 0.f, 0.f};

#pragma unroll 1
    for (int jj = 0; jj < 8; ++jj) {
        int j = jb * 64 + wave * 8 + jj;
        float xj0 = x[j * 3 + 0], xj1 = x[j * 3 + 1], xj2 = x[j * 3 + 2];
        float gsum0 = 0.f, gsum1 = 0.f, gsum2 = 0.f;
#pragma unroll 1
        for (int ic = 0; ic < 4; ++ic) {
            int i = ib * 64 + ic * 16 + lm;  // A row m = lm
            float ft[49];
            pair_features(x[i * 3 + 0] - xj0, x[i * 3 + 1] - xj1, x[i * 3 + 2] - xj2, ft);

            // Build A operands (K=0..63, 49 valid). A half h: Klo=(h<8?h:h+8)(+8 hi-lanes).
            v16h A0, A1;
#pragma unroll
            for (int h = 0; h < 16; ++h) {
                int Klo = (h < 8) ? h : (h + 8);
                float v0a = (Klo < 49) ? ft[Klo] : 0.0f;
                float v0b = (Klo + 8 < 49) ? ft[Klo + 8] : 0.0f;
                float v1a = (Klo + 32 < 49) ? ft[Klo + 32] : 0.0f;
                float v1b = (Klo + 40 < 49) ? ft[Klo + 40] : 0.0f;
                A0[h] = (_Float16)(lhalf ? v0b : v0a);
                A1[h] = (_Float16)(lhalf ? v1b : v1a);
            }

            // Layer 0: h2_1 = tanh(feat @ W2_0 + b)
            v8f c0 = __builtin_amdgcn_wmma_f32_16x16x32_f16(false, A0, false, B0[0][0], (short)0, zero, false, false);
            c0 = __builtin_amdgcn_wmma_f32_16x16x32_f16(false, A1, false, B0[0][1], (short)0, c0, false, false);
            v8f c1 = __builtin_amdgcn_wmma_f32_16x16x32_f16(false, A0, false, B0[1][0], (short)0, zero, false, false);
            c1 = __builtin_amdgcn_wmma_f32_16x16x32_f16(false, A1, false, B0[1][1], (short)0, c1, false, false);
            v8f h2a, h2b;
#pragma unroll
            for (int r = 0; r < 8; ++r) {
                h2a[r] = ftanh(c0[r] + bia0[0]);
                h2b[r] = ftanh(c1[r] + bia0[1]);
            }
            gsum0 += reduce_mn(h2a, h2b, lhalf);

            // Layer 1: h2_2 = tanh(h2_1 @ W2_1 + b) + h2_1  (transpose via LDS)
            lds_store_tile(xT[wave], h2a, h2b, lhalf, lm);
            asm volatile("s_wait_dscnt 0" ::: "memory");
            v16h A = lds_load_A(xT[wave], lhalf, lm);
            c0 = __builtin_amdgcn_wmma_f32_16x16x32_f16(false, A, false, B1[0], (short)0, zero, false, false);
            c1 = __builtin_amdgcn_wmma_f32_16x16x32_f16(false, A, false, B1[1], (short)0, zero, false, false);
#pragma unroll
            for (int r = 0; r < 8; ++r) {
                h2a[r] += ftanh(c0[r] + bia1[0]);
                h2b[r] += ftanh(c1[r] + bia1[1]);
            }
            gsum1 += reduce_mn(h2a, h2b, lhalf);

            // Layer 2: h2_3 = tanh(h2_2 @ W2_2 + b) + h2_2
            lds_store_tile(xT[wave], h2a, h2b, lhalf, lm);
            asm volatile("s_wait_dscnt 0" ::: "memory");
            A = lds_load_A(xT[wave], lhalf, lm);
            c0 = __builtin_amdgcn_wmma_f32_16x16x32_f16(false, A, false, B2[0], (short)0, zero, false, false);
            c1 = __builtin_amdgcn_wmma_f32_16x16x32_f16(false, A, false, B2[1], (short)0, zero, false, false);
#pragma unroll
            for (int r = 0; r < 8; ++r) {
                h2a[r] += ftanh(c0[r] + bia2[0]);
                h2b[r] += ftanh(c1[r] + bia2[1]);
            }
            gsum2 += reduce_mn(h2a, h2b, lhalf);
        }
        int base = (ib * 1024 + j) * 96;  // lane owns feature n = lane
        partial[base + 0 * 32 + lane] = gsum0;
        partial[base + 1 * 32 + lane] = gsum1;
        partial[base + 2 * 32 + lane] = gsum2;
    }
}

// ---------------------------------------------------------------------------
// Reduce 16 i-block partials -> g2 means (deterministic fixed-order sums)
// ---------------------------------------------------------------------------
__global__ void reduce_g2(const float* __restrict__ partial, float* __restrict__ g2m) {
    int idx = blockIdx.x * blockDim.x + threadIdx.x;
    if (idx >= 1024 * 96) return;
    int j = idx / 96, c = idx % 96;
    float s = 0.f;
#pragma unroll
    for (int ibk = 0; ibk < 16; ++ibk) s += partial[(ibk * 1024 + j) * 96 + c];
    g2m[idx] = s * (1.0f / 1024.0f);
}

// ---------------------------------------------------------------------------
// h1 chain: WMMA over M=1024 rows, N=64. One wave per 16-row group (grid 8 x 8 waves).
// ---------------------------------------------------------------------------
__global__ __launch_bounds__(256) void h1_layer0(
    const float* __restrict__ tptr, const float* __restrict__ g2_0,
    const float* __restrict__ W1, const float* __restrict__ b1,
    float* __restrict__ h1out) {
    __shared__ float tfs[17];
    if (threadIdx.x == 0) {
        float t = tptr[0];
        tfs[0] = t;
#pragma unroll
        for (int kf = 1; kf <= 8; ++kf) {
            float w = TWO_PI_F * t * (float)kf;
            tfs[1 + 2 * (kf - 1)] = __cosf(w);
            tfs[2 + 2 * (kf - 1)] = __sinf(w);
        }
    }
    __syncthreads();
    int wave = threadIdx.x >> 5, lane = threadIdx.x & 31;
    int lhalf = lane >> 4, lm = lane & 15;
    int g = blockIdx.x * 8 + wave;
    int row = g * 16 + lm;
    const v8f zero = {0.f, 0.f, 0.f, 0.f, 0.f, 0.f, 0.f, 0.f};

    auto loadf = [&](int K) -> float {  // f0 = [tf(17), tf(17), g2_0(49)], pad to 96
        if (K < 17) return tfs[K];
        if (K < 34) return tfs[K - 17];
        if (K < 83) return g2_0[row * 49 + (K - 34)];
        return 0.0f;
    };
    v8f acc[4] = {zero, zero, zero, zero};
#pragma unroll
    for (int c = 0; c < 3; ++c) {
        v16h A;
#pragma unroll
        for (int h = 0; h < 16; ++h) {
            int Klo = (h < 8) ? h : (h + 8);
            A[h] = (_Float16)loadf(c * 32 + Klo + 8 * lhalf);
        }
#pragma unroll
        for (int t = 0; t < 4; ++t) {
            v16h Bt;
            int n = lm + 16 * t;
#pragma unroll
            for (int h = 0; h < 16; ++h) {
                int K = c * 32 + 8 * lhalf + ((h < 8) ? h : (h + 8));
                Bt[h] = (_Float16)((K < 83) ? W1[K * 64 + n] : 0.0f);
            }
            acc[t] = __builtin_amdgcn_wmma_f32_16x16x32_f16(false, A, false, Bt, (short)0, acc[t], false, false);
        }
    }
#pragma unroll
    for (int t = 0; t < 4; ++t) {
        int n = lm + 16 * t;
        float bb = b1[n];
#pragma unroll
        for (int r = 0; r < 8; ++r) {
            int m = r + 8 * lhalf;
            h1out[(g * 16 + m) * 64 + n] = ftanh(acc[t][r] + bb);
        }
    }
}

__global__ __launch_bounds__(256) void h1_layerd(
    const float* __restrict__ h1in, const float* __restrict__ g1,
    const float* __restrict__ g2m, int doff,
    const float* __restrict__ W1, const float* __restrict__ b1,
    float* __restrict__ h1out) {
    int wave = threadIdx.x >> 5, lane = threadIdx.x & 31;
    int lhalf = lane >> 4, lm = lane & 15;
    int g = blockIdx.x * 8 + wave;
    int row = g * 16 + lm;
    const v8f zero = {0.f, 0.f, 0.f, 0.f, 0.f, 0.f, 0.f, 0.f};

    auto loadf = [&](int K) -> float {  // f = [h1(64), g1(64), g2_d(32)] (K=160 exact)
        if (K < 64) return h1in[row * 64 + K];
        if (K < 128) return g1[K - 64];
        return g2m[row * 96 + doff + (K - 128)];
    };
    v8f acc[4] = {zero, zero, zero, zero};
#pragma unroll
    for (int c = 0; c < 5; ++c) {
        v16h A;
#pragma unroll
        for (int h = 0; h < 16; ++h) {
            int Klo = (h < 8) ? h : (h + 8);
            A[h] = (_Float16)loadf(c * 32 + Klo + 8 * lhalf);
        }
#pragma unroll
        for (int t = 0; t < 4; ++t) {
            v16h Bt;
            int n = lm + 16 * t;
#pragma unroll
            for (int h = 0; h < 16; ++h) {
                int K = c * 32 + 8 * lhalf + ((h < 8) ? h : (h + 8));
                Bt[h] = (_Float16)W1[K * 64 + n];
            }
            acc[t] = __builtin_amdgcn_wmma_f32_16x16x32_f16(false, A, false, Bt, (short)0, acc[t], false, false);
        }
    }
#pragma unroll
    for (int t = 0; t < 4; ++t) {
        int n = lm + 16 * t;
        float bb = b1[n];
#pragma unroll
        for (int r = 0; r < 8; ++r) {
            int m = r + 8 * lhalf;
            int o = (g * 16 + m) * 64 + n;
            h1out[o] = ftanh(acc[t][r] + bb) + h1in[o];  // residual
        }
    }
}

__global__ __launch_bounds__(256) void mean_h1(const float* __restrict__ h1,
                                               float* __restrict__ g1) {
    int tid = threadIdx.x;
    int col = tid & 63, part = tid >> 6;
    float s = 0.f;
    for (int r = part * 256; r < part * 256 + 256; ++r) s += h1[r * 64 + col];
    __shared__ float ps[4][64];
    ps[part][col] = s;
    __syncthreads();
    if (tid < 64)
        g1[tid] = (ps[0][tid] + ps[1][tid] + ps[2][tid] + ps[3][tid]) * (1.0f / 1024.0f);
}

__global__ void final_out(const float* __restrict__ h1, const float* __restrict__ Wf,
                          const float* __restrict__ x, float* __restrict__ out) {
    int row = blockIdx.x * blockDim.x + threadIdx.x;
    if (row >= 1024) return;
    float s0 = 0.f, s1 = 0.f, s2 = 0.f;
#pragma unroll 8
    for (int k = 0; k < 64; ++k) {
        float h = h1[row * 64 + k];
        s0 += h * Wf[k * 3 + 0];
        s1 += h * Wf[k * 3 + 1];
        s2 += h * Wf[k * 3 + 2];
    }
    float sd[3] = {s0, s1, s2};
#pragma unroll
    for (int d = 0; d < 3; ++d) {
        float fx = 2.0f * x[row * 3 + d];
        fx = fminf(fmaxf(fx, -10.0f), 10.0f);
        out[row * 3 + d] = -sd[d] * fx;
    }
}

extern "C" void kernel_launch(void* const* d_in, const int* in_sizes, int n_in,
                              void* d_out, int out_size, void* d_ws, size_t ws_size,
                              hipStream_t stream) {
    const float* x  = (const float*)d_in[0];
    const float* t  = (const float*)d_in[1];
    const float* W1_0 = (const float*)d_in[2];
    const float* b1_0 = (const float*)d_in[3];
    const float* W1_1 = (const float*)d_in[4];
    const float* b1_1 = (const float*)d_in[5];
    const float* W1_2 = (const float*)d_in[6];
    const float* b1_2 = (const float*)d_in[7];
    const float* W1_3 = (const float*)d_in[8];
    const float* b1_3 = (const float*)d_in[9];
    const float* W2_0 = (const float*)d_in[10];
    const float* b2_0 = (const float*)d_in[11];
    const float* W2_1 = (const float*)d_in[12];
    const float* b2_1 = (const float*)d_in[13];
    const float* W2_2 = (const float*)d_in[14];
    const float* b2_2 = (const float*)d_in[15];
    const float* Wf   = (const float*)d_in[16];
    float* out = (float*)d_out;

    float* ws      = (float*)d_ws;
    float* partial = ws;                     // 1,572,864 floats
    float* g2_0    = partial + 16 * 1024 * 96;
    float* g2m     = g2_0 + 1024 * 49;
    float* g1      = g2m + 1024 * 96;
    float* h1a     = g1 + 64;
    float* h1b     = h1a + 1024 * 64;

    g2_0_kernel<<<1024, 256, 0, stream>>>(x, g2_0);
    h2_main<<<dim3(16, 16), 256, 0, stream>>>(x, W2_0, b2_0, W2_1, b2_1, W2_2, b2_2, partial);
    reduce_g2<<<384, 256, 0, stream>>>(partial, g2m);

    h1_layer0<<<8, 256, 0, stream>>>(t, g2_0, W1_0, b1_0, h1a);
    mean_h1<<<1, 256, 0, stream>>>(h1a, g1);
    h1_layerd<<<8, 256, 0, stream>>>(h1a, g1, g2m, 0,  W1_1, b1_1, h1b);
    mean_h1<<<1, 256, 0, stream>>>(h1b, g1);
    h1_layerd<<<8, 256, 0, stream>>>(h1b, g1, g2m, 32, W1_2, b1_2, h1a);
    mean_h1<<<1, 256, 0, stream>>>(h1a, g1);
    h1_layerd<<<8, 256, 0, stream>>>(h1a, g1, g2m, 64, W1_3, b1_3, h1b);
    final_out<<<4, 256, 0, stream>>>(h1b, Wf, x, out);
}
